// Hourglass_Concat_64905545777568
// MI455X (gfx1250) — matrix operand order repaired and checked by reading, not compiled, not run on hardware
//
#include <hip/hip_runtime.h>
#include <math.h>

typedef _Float16 half_t;
typedef __attribute__((ext_vector_type(16))) _Float16 v16h;
typedef __attribute__((ext_vector_type(8)))  float    v8f;

#define NBATCH 4
#define NS 4096
#define NB 8192

// ---------------------------------------------------------------------------
// Batched WMMA GEMM: C[M,N] (+)= A[M,K] * B[K,N]  (+ bias[N])
// f32 inputs staged branchlessly (clamp+mask) into *fragment-ready* f16 LDS
// layouts so each lane's WMMA fragment is a single contiguous 32B LDS read
// (2x ds_load_b128), then v_wmma_f32_16x16x32_f16 with f32 accumulate.
// Block = 128 threads (4 waves). Wave: 16x32 of C (one A frag, two B frags,
// two WMMA per K-step). Block tile: 64(M) x 32(N), K stepped by 32.
//
// Afr[p][m][j] = A[m][ k0 + 16p + j ]  (contiguous k half per lane half,
//   p = lane>=16, j = fragment elem; matches the B fragment k order)
// Bfr[p][n][h] = B[k0 + 16p + h][bn + n]  (B fragment: k half per lane half)
// ---------------------------------------------------------------------------
__global__ __launch_bounds__(128)
void gemm_wmma_kernel(const float* __restrict__ A, int lda, long long sA,
                      const float* __restrict__ Bm, int ldb, long long sB,
                      float* __restrict__ C, int ldc, long long sC,
                      const float* __restrict__ bias,
                      int M, int N, int K, int acc)
{
    __shared__ __align__(16) half_t Afr[2][64][16];
    __shared__ __align__(16) half_t Bfr[2][32][16];

    const int tid  = threadIdx.x;
    const int wave = tid >> 5;
    const int lane = tid & 31;
    const int bm   = blockIdx.y * 64;
    const int bn   = blockIdx.x * 32;
    const int bz   = blockIdx.z;

    const float* Ab = A  + (long long)bz * sA;
    const float* Bb = Bm + (long long)bz * sB;
    float*       Cb = C  + (long long)bz * sC;

    // A staging: thread -> (row ar, lane-half ap). One contiguous 16-float run.
    const int ar = tid >> 1;
    const int ap = tid & 1;
    const int agm = bm + ar;
    const float* arow = Ab + (long long)(agm < M ? agm : (M - 1)) * lda;
    const float  arm  = (agm < M) ? 1.f : 0.f;

    // B staging: thread -> (k-row bk, 8-wide col chunk bc0). Contiguous loads.
    const int bk  = tid >> 2;
    const int bc0 = (tid & 3) * 8;
    const int bp  = bk >> 4;
    const int bh  = bk & 15;

    v8f c0 = {}, c1 = {};
    const int kt = (K + 31) >> 5;
    for (int t = 0; t < kt; ++t) {
        const int k0 = t << 5;
        // ---- stage A (fragment-ready, branchless) ----
        #pragma unroll
        for (int j = 0; j < 16; ++j) {
            int kk  = 16 * ap + j;
            int gk  = k0 + kk;
            int gkc = gk < K ? gk : (K - 1);
            float v = arow[gkc] * ((gk < K) ? arm : 0.f);
            Afr[ap][ar][j] = (half_t)v;
        }
        // ---- stage B (fragment-ready, branchless) ----
        {
            int gk  = k0 + bk;
            int gkc = gk < K ? gk : (K - 1);
            const float* brow = Bb + (long long)gkc * ldb;
            float bkm = (gk < K) ? 1.f : 0.f;
            #pragma unroll
            for (int i = 0; i < 8; ++i) {
                int gn  = bn + bc0 + i;
                int gnc = gn < N ? gn : (N - 1);
                float v = brow[gnc] * ((gn < N) ? bkm : 0.f);
                Bfr[bp][bc0 + i][bh] = (half_t)v;
            }
        }
        __syncthreads();

        // ---- fragments: contiguous 32B LDS reads per lane ----
        const int lp = lane >> 4;
        const int am = wave * 16 + (lane & 15);
        v16h afrag  = *(const v16h*)(&Afr[lp][am][0]);
        v16h bfrag0 = *(const v16h*)(&Bfr[lp][lane & 15][0]);
        v16h bfrag1 = *(const v16h*)(&Bfr[lp][16 + (lane & 15)][0]);

        c0 = __builtin_amdgcn_wmma_f32_16x16x32_f16(
                false, afrag, false, bfrag0, (short)0, c0, false, false);
        c1 = __builtin_amdgcn_wmma_f32_16x16x32_f16(
                false, afrag, false, bfrag1, (short)0, c1, false, false);
        __syncthreads();
    }

    // ---- epilogue: C layout lane=(n, m-half), vgpr v -> m = v + 8*(lane>=16) ----
    const int mo  = (lane < 16) ? 0 : 8;
    const int nlo = lane & 15;
    #pragma unroll
    for (int vv = 0; vv < 8; ++vv) {
        int gm = bm + wave * 16 + mo + vv;
        if (gm < M) {
            long long rowb = (long long)gm * ldc;
            int gn0 = bn + nlo;
            if (gn0 < N) {
                float o = c0[vv];
                if (acc)  o += Cb[rowb + gn0];
                if (bias) o += bias[gn0];
                Cb[rowb + gn0] = o;
            }
            int gn1 = bn + 16 + nlo;
            if (gn1 < N) {
                float o = c1[vv];
                if (acc)  o += Cb[rowb + gn1];
                if (bias) o += bias[gn1];
                Cb[rowb + gn1] = o;
            }
        }
    }
}

// ---------------------------------------------------------------------------
// z [4,64] x dec_w[524288,64]^T + dec_b, ReLU -> out [4,4096,128] packed.
// Pure bandwidth op (134 MB weight read): plain VALU kernel.
// ---------------------------------------------------------------------------
__global__ void decode_relu_kernel(const float* __restrict__ z,
                                   const float* __restrict__ W,
                                   const float* __restrict__ bias,
                                   float* __restrict__ out, int total)
{
    int i = blockIdx.x * blockDim.x + threadIdx.x;
    if (i >= total) return;
    int nf = i % (NS * 128);
    int b  = i / (NS * 128);
    const float* w  = W + (long long)nf * 64;
    const float* zz = z + b * 64;
    float s = bias[nf];
    #pragma unroll 8
    for (int k = 0; k < 64; ++k) s += zz[k] * w[k];
    out[i] = s > 0.f ? s : 0.f;
}

__global__ void zero_kernel(float* __restrict__ p, long long n)
{
    long long i = (long long)blockIdx.x * blockDim.x + threadIdx.x;
    if (i < n) p[i] = 0.f;
}

__global__ void negcopy_kernel(float* __restrict__ dst, const float* __restrict__ src,
                               long long n)
{
    long long i = (long long)blockIdx.x * blockDim.x + threadIdx.x;
    if (i < n) dst[i] = -src[i];
}

__global__ void deg_kernel(const int* __restrict__ row, int E, float* __restrict__ deg)
{
    int e = blockIdx.x * blockDim.x + threadIdx.x;
    if (e < E)
        __hip_atomic_fetch_add(&deg[row[e]], 1.0f, __ATOMIC_RELAXED,
                               __HIP_MEMORY_SCOPE_AGENT);
}

__global__ void dis_kernel(float* __restrict__ deg, int n)
{
    int i = blockIdx.x * blockDim.x + threadIdx.x;
    if (i < n) {
        float d = deg[i];
        deg[i] = d > 0.f ? rsqrtf(d) : 0.f;
    }
}

// dst[:, row, f] += scale * (-(dis[row]*dis[col])) * src[:, col, f]   (packed ld=C)
__global__ void scatter_kernel(float* __restrict__ dst, const float* __restrict__ src,
                               const int* __restrict__ row, const int* __restrict__ col,
                               const float* __restrict__ dis,
                               int E, int N, int C, float scale)
{
    long long i = (long long)blockIdx.x * blockDim.x + threadIdx.x;
    long long total = (long long)E * C;
    if (i >= total) return;
    int e = (int)(i / C), f = (int)(i % C);
    int r = row[e], cl = col[e];
    float w = -(dis[r] * dis[cl]) * scale;
    long long bs = (long long)N * C;
    long long di = (long long)r  * C + f;
    long long si = (long long)cl * C + f;
    #pragma unroll
    for (int b = 0; b < NBATCH; ++b)
        __hip_atomic_fetch_add(&dst[b * bs + di], w * src[b * bs + si],
                               __ATOMIC_RELAXED, __HIP_MEMORY_SCOPE_AGENT);
}

// InstanceNorm over last dim (C cols of a row with stride ld) + ReLU, in place.
// One wave per row.
__global__ __launch_bounds__(128)
void inorm_relu_kernel(float* __restrict__ x, int ld, int C, int rows)
{
    int r = blockIdx.x * 4 + (threadIdx.x >> 5);
    int lane = threadIdx.x & 31;
    if (r >= rows) return;
    float* p = x + (long long)r * ld;
    float s = 0.f, s2 = 0.f;
    for (int f = lane; f < C; f += 32) { float v = p[f]; s += v; s2 += v * v; }
    #pragma unroll
    for (int m = 16; m >= 1; m >>= 1) {
        s  += __shfl_xor(s,  m, 32);
        s2 += __shfl_xor(s2, m, 32);
    }
    float mean = s / C;
    float var  = s2 / C - mean * mean;
    float inv  = rsqrtf(var + 1e-5f);
    for (int f = lane; f < C; f += 32) {
        float v = (p[f] - mean) * inv;
        p[f] = v > 0.f ? v : 0.f;
    }
}

// pos = x[:, :C] @ Wo[C,2]; write to d_out slice and (optionally) concat cols.
__global__ void pos_head_kernel(const float* __restrict__ x, int ld, int C,
                                const float* __restrict__ Wo,
                                float* __restrict__ outpos,
                                float* concatDst, int cld, int coff, int rows)
{
    int r = blockIdx.x * blockDim.x + threadIdx.x;
    if (r >= rows) return;
    const float* p = x + (long long)r * ld;
    float a0 = 0.f, a1 = 0.f;
    for (int f = 0; f < C; ++f) {
        float v = p[f];
        a0 += v * Wo[2 * f];
        a1 += v * Wo[2 * f + 1];
    }
    outpos[2 * r]     = a0;
    outpos[2 * r + 1] = a1;
    if (concatDst) {
        concatDst[(long long)r * cld + coff]     = a0;
        concatDst[(long long)r * cld + coff + 1] = a1;
    }
}

// roi_align(1x1, aligned, 3x3 grid) around pos*W on conv [B,Fc,H,W].
__global__ void lookup_kernel(const float* __restrict__ pos,
                              const float* __restrict__ conv,
                              float* __restrict__ dst, int ld, int coff,
                              int N, int Fc, int H, int W)
{
    int i = blockIdx.x * blockDim.x + threadIdx.x;
    int total = NBATCH * N * Fc;
    if (i >= total) return;
    int f = i % Fc;
    int n = (i / Fc) % N;
    int b = i / (Fc * N);
    long long pr = (long long)b * N + n;
    float px = pos[2 * pr], py = pos[2 * pr + 1];
    float h  = (float)W;
    float sx = px * h - 1.5f;
    float sy = py * h - 1.5f;
    const float* img = conv + ((long long)b * Fc + f) * (long long)H * W;
    float acc = 0.f;
    #pragma unroll
    for (int j = 0; j < 3; ++j) {
        float y = sy + 0.5f + (float)j;
        #pragma unroll
        for (int ii = 0; ii < 3; ++ii) {
            float x = sx + 0.5f + (float)ii;
            bool valid = (y >= -1.f) && (y <= (float)H) && (x >= -1.f) && (x <= (float)W);
            float yc = fminf(fmaxf(y, 0.f), (float)(H - 1));
            float xc = fminf(fmaxf(x, 0.f), (float)(W - 1));
            float y0 = floorf(yc), x0 = floorf(xc);
            float ly = yc - y0, lx = xc - x0;
            int y0i = (int)y0, x0i = (int)x0;
            int y1i = y0i + 1 < H ? y0i + 1 : H - 1;
            int x1i = x0i + 1 < W ? x0i + 1 : W - 1;
            float v = img[y0i * W + x0i] * (1.f - ly) * (1.f - lx)
                    + img[y0i * W + x1i] * (1.f - ly) * lx
                    + img[y1i * W + x0i] * ly * (1.f - lx)
                    + img[y1i * W + x1i] * ly * lx;
            acc += valid ? v : 0.f;
        }
    }
    dst[pr * ld + coff + f] = acc * (1.0f / 9.0f);
}

// ---------------------------------------------------------------------------
// Host-side helpers (all launches on stream; no allocation, graph-safe)
// ---------------------------------------------------------------------------
static inline void launch_gemm(const float* A, int lda, long long sA,
                               const float* Bm, int ldb, long long sB,
                               float* C, int ldc, long long sC,
                               const float* bias, int M, int N, int K, int acc,
                               int batches, hipStream_t stream)
{
    dim3 grid((N + 31) / 32, (M + 63) / 64, batches);
    gemm_wmma_kernel<<<grid, dim3(128), 0, stream>>>(A, lda, sA, Bm, ldb, sB,
                                                     C, ldc, sC, bias, M, N, K, acc);
}

static void run_cheb6(float* xin, int Cin, int Nn, const int* adj, int E,
                      const float* Wup, const float* bias, int Fout,
                      float* dst, int ldc,
                      float* s1, float* s2, float* deg, hipStream_t stream)
{
    const long long elems = (long long)NBATCH * Nn * Cin;
    const int M = NBATCH * Nn;
    const int* row = adj;
    const int* col = adj + E;

    // k = 0: out = Tx0 @ W0 + bias
    launch_gemm(xin, Cin, 0, Wup, Fout, 0, dst, ldc, 0, bias, M, Fout, Cin, 0, 1, stream);

    // normalized -D^-1/2 A D^-1/2 weights
    zero_kernel<<<(Nn + 255) / 256, 256, 0, stream>>>(deg, Nn);
    deg_kernel<<<(E + 255) / 256, 256, 0, stream>>>(row, E, deg);
    dis_kernel<<<(Nn + 255) / 256, 256, 0, stream>>>(deg, Nn);

    long long sc_total = (long long)E * Cin;
    int zb = (int)((elems + 255) / 256);
    int sb = (int)((sc_total + 255) / 256);

    // k = 1: Tx1 = L(Tx0)
    zero_kernel<<<zb, 256, 0, stream>>>(s1, elems);
    scatter_kernel<<<sb, 256, 0, stream>>>(s1, xin, row, col, deg, E, Nn, Cin, 1.0f);
    launch_gemm(s1, Cin, 0, Wup + (long long)1 * Cin * Fout, Fout, 0,
                dst, ldc, 0, nullptr, M, Fout, Cin, 1, 1, stream);

    // k = 2..5: Tx2 = 2 L(Tx1) - Tx0
    float* p0 = xin;
    float* p1 = s1;
    float* p2 = s2;
    for (int k = 2; k < 6; ++k) {
        negcopy_kernel<<<zb, 256, 0, stream>>>(p2, p0, elems);
        scatter_kernel<<<sb, 256, 0, stream>>>(p2, p1, row, col, deg, E, Nn, Cin, 2.0f);
        launch_gemm(p2, Cin, 0, Wup + (long long)k * Cin * Fout, Fout, 0,
                    dst, ldc, 0, nullptr, M, Fout, Cin, 1, 1, stream);
        float* t = p0; p0 = p1; p1 = p2; p2 = t;
    }
}

extern "C" void kernel_launch(void* const* d_in, const int* in_sizes, int n_in,
                              void* d_out, int out_size, void* d_ws, size_t ws_size,
                              hipStream_t stream)
{
    (void)in_sizes; (void)n_in; (void)out_size; (void)ws_size;

    const float* z      = (const float*)d_in[0];
    const float* conv1  = (const float*)d_in[1];
    const float* conv2  = (const float*)d_in[2];
    const float* conv3  = (const float*)d_in[3];
    const float* conv4  = (const float*)d_in[4];
    const float* conv5  = (const float*)d_in[5];
    const float* conv6  = (const float*)d_in[6];
    const float* up_mat = (const float*)d_in[7];
    const float* dec_w  = (const float*)d_in[8];
    const float* dec_b  = (const float*)d_in[9];
    const float* w_up6  = (const float*)d_in[10];
    const float* b_up6  = (const float*)d_in[11];
    const float* w_up5  = (const float*)d_in[12];
    const float* b_up5  = (const float*)d_in[13];
    const float* w_up4  = (const float*)d_in[14];
    const float* b_up4  = (const float*)d_in[15];
    const float* w_up3  = (const float*)d_in[16];
    const float* b_up3  = (const float*)d_in[17];
    const float* w_up2  = (const float*)d_in[18];
    const float* b_up2  = (const float*)d_in[19];
    const float* w_up1  = (const float*)d_in[20];
    const float* b_up1  = (const float*)d_in[21];
    const float* w_out6 = (const float*)d_in[22];
    const float* w_out5 = (const float*)d_in[23];
    const float* w_out4 = (const float*)d_in[24];
    const float* w_out3 = (const float*)d_in[25];
    const float* w_out2 = (const float*)d_in[26];
    const float* w_out1 = (const float*)d_in[27];
    const float* w_out0 = (const float*)d_in[28];
    const int*   adj5   = (const int*)d_in[29];
    const int*   adj4   = (const int*)d_in[30];
    const int*   adj3   = (const int*)d_in[31];
    const int*   adj2   = (const int*)d_in[32];
    const int*   adj1   = (const int*)d_in[33];

    const int ES = NS * 16, EB = NB * 16;

    float* ws = (float*)d_ws;
    const long long BUF = (long long)NBATCH * NB * 130; // 4,259,840 floats
    float* bufA = ws;
    float* bufB = bufA + BUF;
    float* s1   = bufB + BUF;
    float* s2   = s1 + BUF;
    float* deg  = s2 + BUF;

    float* out  = (float*)d_out;
    float* pos7 = out;            // [4,8192,2]
    float* pos6 = out + 65536;
    float* pos5 = out + 131072;
    float* pos4 = out + 196608;
    float* pos3 = out + 262144;   // [4,4096,2]
    float* pos2 = out + 294912;
    float* pos1 = out + 327680;

    // Stage 0: decode + ReLU -> bufA [4,4096,128] packed
    {
        int total = NBATCH * NS * 128;
        decode_relu_kernel<<<(total + 255) / 256, 256, 0, stream>>>(z, dec_w, dec_b,
                                                                    bufA, total);
    }

    const int rNS = NBATCH * NS, rNB = NBATCH * NB;

    // L6: Cin=128 -> Fout=128, conv6 (Fc=128, 8x8), next C=258 (bufB)
    run_cheb6(bufA, 128, NS, adj5, ES, w_up6, b_up6, 128, bufB, 258, s1, s2, deg, stream);
    inorm_relu_kernel<<<(rNS + 3) / 4, 128, 0, stream>>>(bufB, 258, 128, rNS);
    pos_head_kernel<<<(rNS + 255) / 256, 256, 0, stream>>>(bufB, 258, 128, w_out6,
                                                           pos1, bufB, 258, 256, rNS);
    lookup_kernel<<<(NBATCH * NS * 128 + 63) / 64, 64, 0, stream>>>(pos1, conv6, bufB,
                                                                    258, 128, NS, 128, 8, 8);

    // L5: Cin=258 -> Fout=128, conv5 (128, 16x16), next C=258 (bufA)
    run_cheb6(bufB, 258, NS, adj4, ES, w_up5, b_up5, 128, bufA, 258, s1, s2, deg, stream);
    inorm_relu_kernel<<<(rNS + 3) / 4, 128, 0, stream>>>(bufA, 258, 128, rNS);
    pos_head_kernel<<<(rNS + 255) / 256, 256, 0, stream>>>(bufA, 258, 128, w_out5,
                                                           pos2, bufA, 258, 256, rNS);
    lookup_kernel<<<(NBATCH * NS * 128 + 63) / 64, 64, 0, stream>>>(pos2, conv5, bufA,
                                                                    258, 128, NS, 128, 16, 16);

    // L4: Cin=258 -> Fout=64, conv4 (64, 32x32), next C=130 (bufB)
    run_cheb6(bufA, 258, NS, adj3, ES, w_up4, b_up4, 64, bufB, 130, s1, s2, deg, stream);
    inorm_relu_kernel<<<(rNS + 3) / 4, 128, 0, stream>>>(bufB, 130, 64, rNS);
    pos_head_kernel<<<(rNS + 255) / 256, 256, 0, stream>>>(bufB, 130, 64, w_out4,
                                                           pos3, bufB, 130, 128, rNS);
    lookup_kernel<<<(NBATCH * NS * 64 + 63) / 64, 64, 0, stream>>>(pos3, conv4, bufB,
                                                                   130, 64, NS, 64, 32, 32);

    // Upsample: bufA[b] = up_mat[8192x4096] @ bufB[b][4096x130]  (batched WMMA GEMM)
    launch_gemm(up_mat, NS, 0,
                bufB, 130, (long long)NS * 130,
                bufA, 130, (long long)NB * 130,
                nullptr, NB, 130, NS, 0, NBATCH, stream);

    // L3: N=8192, adj2, Cin=130 -> Fout=64, conv3 (32, 64x64), next C=98 (bufB)
    run_cheb6(bufA, 130, NB, adj2, EB, w_up3, b_up3, 64, bufB, 98, s1, s2, deg, stream);
    inorm_relu_kernel<<<(rNB + 3) / 4, 128, 0, stream>>>(bufB, 98, 64, rNB);
    pos_head_kernel<<<(rNB + 255) / 256, 256, 0, stream>>>(bufB, 98, 64, w_out3,
                                                           pos4, bufB, 98, 96, rNB);
    lookup_kernel<<<(NBATCH * NB * 32 + 63) / 64, 64, 0, stream>>>(pos4, conv3, bufB,
                                                                   98, 64, NB, 32, 64, 64);

    // L2: adj1, Cin=98 -> Fout=64, conv2 (16, 128x128), next C=82 (bufA)
    run_cheb6(bufB, 98, NB, adj1, EB, w_up2, b_up2, 64, bufA, 82, s1, s2, deg, stream);
    inorm_relu_kernel<<<(rNB + 3) / 4, 128, 0, stream>>>(bufA, 82, 64, rNB);
    pos_head_kernel<<<(rNB + 255) / 256, 256, 0, stream>>>(bufA, 82, 64, w_out2,
                                                           pos5, bufA, 82, 80, rNB);
    lookup_kernel<<<(NBATCH * NB * 16 + 63) / 64, 64, 0, stream>>>(pos5, conv2, bufA,
                                                                   82, 64, NB, 16, 128, 128);

    // L1: adj1, Cin=82 -> Fout=64, conv1 (8, 256x256), next C=74 (bufB)
    run_cheb6(bufA, 82, NB, adj1, EB, w_up1, b_up1, 64, bufB, 74, s1, s2, deg, stream);
    inorm_relu_kernel<<<(rNB + 3) / 4, 128, 0, stream>>>(bufB, 74, 64, rNB);
    pos_head_kernel<<<(rNB + 255) / 256, 256, 0, stream>>>(bufB, 74, 64, w_out1,
                                                           pos6, bufB, 74, 72, rNB);
    lookup_kernel<<<(NBATCH * NB * 8 + 63) / 64, 64, 0, stream>>>(pos6, conv1, bufB,
                                                                  74, 64, NB, 8, 256, 256);

    // Final head: pos7 = x[74] @ w_out0
    pos_head_kernel<<<(rNB + 255) / 256, 256, 0, stream>>>(bufB, 74, 74, w_out0,
                                                           pos7, nullptr, 0, 0, rNB);
}
